// DCN_sep_50921132261607
// MI455X (gfx1250) — matrix operand order repaired and checked by reading, not compiled, not run on hardware
//
#include <hip/hip_runtime.h>
#include <math.h>

typedef __attribute__((ext_vector_type(16))) _Float16 v16h;
typedef __attribute__((ext_vector_type(8)))  float    v8f;
typedef __attribute__((ext_vector_type(4)))  float    v4f;

#define Hh   128
#define Ww   128
#define HW   (Hh*Ww)
#define NB   8
#define CINc 64
#define COUTc 64
#define COM  216          // G*3*K*K = 8*27
#define NT1  14           // ceil(216/16)
#define NT2  4            // 64/16
#define KSTEPS 18         // 576/32

// ---------------------------------------------------------------------------
// Kernel T: NCHW -> NHWC (64-channel) transpose for x and fea via LDS tiles.
// ---------------------------------------------------------------------------
__global__ void __launch_bounds__(256)
nchw_to_nhwc64(const float* __restrict__ x, const float* __restrict__ fea,
               float* __restrict__ xT, float* __restrict__ feaT) {
    __shared__ float tile[64][65];
    const float* src = blockIdx.y ? fea : x;
    float*       dst = blockIdx.y ? feaT : xT;
    const int b  = blockIdx.x >> 8;
    const int p0 = (blockIdx.x & 255) * 64;
    const int pc = threadIdx.x & 63;
    const int r0 = threadIdx.x >> 6;
    const size_t inb = (size_t)b * 64 * HW + p0;
    #pragma unroll
    for (int c = r0; c < 64; c += 4)
        tile[c][pc] = src[inb + (size_t)c * HW + pc];
    __syncthreads();
    const size_t outb = ((size_t)b * HW + p0) * 64;
    #pragma unroll
    for (int p = r0; p < 64; p += 4)
        dst[outb + (size_t)p * 64 + pc] = tile[pc][p];
}

// ---------------------------------------------------------------------------
// Kernel P: pack weights (f32) into f16 per-lane WMMA B-fragment order.
// K-order: kflat = kk*64 + cin.
// ---------------------------------------------------------------------------
__global__ void __launch_bounds__(256)
pack_wmma_b(const float* __restrict__ w_om, const float* __restrict__ wgt,
            _Float16* __restrict__ wB1, _Float16* __restrict__ wB2) {
    const int N1 = NT1 * KSTEPS * 32;
    const int NT = N1 + NT2 * KSTEPS * 32;
    int tid = blockIdx.x * blockDim.x + threadIdx.x;
    if (tid >= NT) return;
    const float* src; _Float16* dst; int nch; int idx;
    if (tid < N1) { src = w_om; dst = wB1; nch = COM;   idx = tid; }
    else          { src = wgt;  dst = wB2; nch = COUTc; idx = tid - N1; }
    const int ntile = idx / (KSTEPS * 32);
    const int rem   = idx % (KSTEPS * 32);
    const int t     = rem >> 5;
    const int lane  = rem & 31;
    const int n     = lane & 15;
    const int kbB   = (lane & 16) ? 8 : 0;
    const int kk    = t >> 1;
    const int chalf = (t & 1) << 5;
    const int oc    = ntile * 16 + n;
    _Float16* o = dst + (size_t)idx * 16;
    #pragma unroll
    for (int e = 0; e < 16; ++e) {
        int c = chalf + kbB + e + (e & 8);
        float w = (oc < nch) ? src[((size_t)oc * 64 + c) * 9 + kk] : 0.f;
        o[e] = (_Float16)w;
    }
}

// ---------------------------------------------------------------------------
// Kernel 1: offset/mask conv as implicit GEMM. Each wave: 16 pixels x 2
// N-tiles (32 out channels), K = 576 in 18 steps. A fragment built once per
// K-step from contiguous NHWC runs (b128 loads), reused for both N-tiles.
// ---------------------------------------------------------------------------
__global__ void __launch_bounds__(128)
offset_conv_wmma(const float* __restrict__ feaT, const float* __restrict__ b_om,
                 const _Float16* __restrict__ wB1, float* __restrict__ om) {
    const int lane   = threadIdx.x & 31;
    const int wave   = threadIdx.x >> 5;
    const int mt     = blockIdx.x * 4 + wave;
    const int ntile0 = blockIdx.y * 2;           // 0,2,...,12
    const int b  = mt >> 10;
    const int y  = (mt >> 3) & 127;
    const int xt = mt & 7;
    const int x  = xt * 16 + (lane & 15);
    const int kbA = (lane & 16) ? 8 : 0;

    v8f acc[2] = {{}, {}};
    for (int t = 0; t < KSTEPS; ++t) {
        const int kk = t >> 1, chalf = (t & 1) << 5;
        const int ky = (kk >= 6) ? 2 : ((kk >= 3) ? 1 : 0);
        const int kx = kk - 3 * ky;
        const int yy = y + ky - 1, xx = x + kx - 1;
        v16h a;
        if ((unsigned)yy < (unsigned)Hh && (unsigned)xx < (unsigned)Ww) {
            const v4f* q = (const v4f*)(feaT + ((size_t)b * HW + (size_t)yy * Ww + xx) * 64
                                             + chalf + kbA);
            const v4f r0 = q[0], r1 = q[1], r2 = q[4], r3 = q[5];
            #pragma unroll
            for (int j = 0; j < 4; ++j) {
                a[j]      = (_Float16)r0[j];
                a[4 + j]  = (_Float16)r1[j];
                a[8 + j]  = (_Float16)r2[j];
                a[12 + j] = (_Float16)r3[j];
            }
        } else {
            #pragma unroll
            for (int i = 0; i < 16; ++i) a[i] = (_Float16)0.f;
        }
        #pragma unroll
        for (int nt = 0; nt < 2; ++nt) {
            v16h bm = *(const v16h*)(wB1 +
                (((size_t)(ntile0 + nt) * KSTEPS + t) * 32 + lane) * 16);
            acc[nt] = __builtin_amdgcn_wmma_f32_16x16x32_f16(
                          false, a, false, bm, (short)0, acc[nt], false, false);
        }
    }
    // epilogue: bias (+ sigmoid for mask channels), slot-remapped store
    const int n  = lane & 15;
    const int mb = (lane & 16) ? 8 : 0;
    #pragma unroll
    for (int nt = 0; nt < 2; ++nt) {
        const int oc = (ntile0 + nt) * 16 + n;
        if (oc < COM) {
            const float bias = b_om[oc];
            int slot; bool sig;
            if (oc < 144) { int g = oc / 18, r = oc % 18;
                            slot = (g * 9 + (r >> 1)) * 3 + (r & 1); sig = false; }
            else          { slot = (oc - 144) * 3 + 2;               sig = true;  }
            #pragma unroll
            for (int r = 0; r < 8; ++r) {
                const int xo = xt * 16 + mb + r;
                float v = acc[nt][r] + bias;
                if (sig) v = 1.f / (1.f + __expf(-v));
                om[((size_t)b * HW + (size_t)y * Ww + xo) * COM + slot] = v;
            }
        }
    }
}

// ---------------------------------------------------------------------------
// Kernel 2: modulated deformable conv as implicit GEMM. Each wave: 16 pixels
// x ALL 4 N-tiles (full 64 out channels). The expensive bilinear-gathered A
// fragment is built once per K-step and feeds 4 WMMAs. Sampling uses packed
// v4f (b128) loads from NHWC x; bilinear blend in packed f32.
// ---------------------------------------------------------------------------
__global__ void __launch_bounds__(128)
deform_conv_wmma(const float* __restrict__ xT, const float* __restrict__ om,
                 const float* __restrict__ bias, const _Float16* __restrict__ wB2,
                 float* __restrict__ out) {
    const int lane = threadIdx.x & 31;
    const int wave = threadIdx.x >> 5;
    const int mt   = blockIdx.x * 4 + wave;
    const int b  = mt >> 10;
    const int y  = (mt >> 3) & 127;
    const int xt = mt & 7;
    const int x  = xt * 16 + (lane & 15);
    const int kbA = (lane & 16) ? 8 : 0;

    const float* omp = om + ((size_t)b * HW + (size_t)y * Ww + x) * COM;
    const float* xb  = xT + (size_t)b * HW * 64;

    v8f acc[4] = {{}, {}, {}, {}};
    for (int t = 0; t < KSTEPS; ++t) {
        const int kk = t >> 1, chalf = (t & 1) << 5;
        const int ky = (kk >= 6) ? 2 : ((kk >= 3) ? 1 : 0);
        const int kx = kk - 3 * ky;
        v16h a;
        #pragma unroll
        for (int h = 0; h < 2; ++h) {
            const int cbase = chalf + kbA + h * 16;   // 8-aligned channel base
            const int g     = cbase >> 3;
            const int sb    = (g * 9 + kk) * 3;       // dy,dx,mask adjacent
            const float dy  = omp[sb], dx = omp[sb + 1], msk = omp[sb + 2];
            const float py  = dy + (float)(y - 1 + ky);
            const float px  = dx + (float)(x - 1 + kx);
            const float y0f = floorf(py), x0f = floorf(px);
            const int   y0  = (int)y0f,   x0  = (int)x0f;
            const float ly  = py - y0f,   lx  = px - x0f;
            const float hy  = 1.f - ly,   hx  = 1.f - lx;
            const bool vy0 = (unsigned)y0       < (unsigned)Hh;
            const bool vy1 = (unsigned)(y0 + 1) < (unsigned)Hh;
            const bool vx0 = (unsigned)x0       < (unsigned)Ww;
            const bool vx1 = (unsigned)(x0 + 1) < (unsigned)Ww;
            const float w00 = hy * hx * (float)(vy0 && vx0);
            const float w01 = hy * lx * (float)(vy0 && vx1);
            const float w10 = ly * hx * (float)(vy1 && vx0);
            const float w11 = ly * lx * (float)(vy1 && vx1);
            const int yi0 = min(max(y0, 0), Hh - 1), yi1 = min(max(y0 + 1, 0), Hh - 1);
            const int xi0 = min(max(x0, 0), Ww - 1), xi1 = min(max(x0 + 1, 0), Ww - 1);
            const v4f* q00 = (const v4f*)(xb + ((size_t)yi0 * Ww + xi0) * 64 + cbase);
            const v4f* q01 = (const v4f*)(xb + ((size_t)yi0 * Ww + xi1) * 64 + cbase);
            const v4f* q10 = (const v4f*)(xb + ((size_t)yi1 * Ww + xi0) * 64 + cbase);
            const v4f* q11 = (const v4f*)(xb + ((size_t)yi1 * Ww + xi1) * 64 + cbase);
            v4f lo = w00 * q00[0] + w01 * q01[0] + w10 * q10[0] + w11 * q11[0];
            v4f hi = w00 * q00[1] + w01 * q01[1] + w10 * q10[1] + w11 * q11[1];
            lo *= msk; hi *= msk;
            #pragma unroll
            for (int j = 0; j < 4; ++j) {
                a[h * 8 + j]     = (_Float16)lo[j];
                a[h * 8 + 4 + j] = (_Float16)hi[j];
            }
        }
        #pragma unroll
        for (int nt = 0; nt < 4; ++nt) {
            v16h bm = *(const v16h*)(wB2 +
                (((size_t)nt * KSTEPS + t) * 32 + lane) * 16);
            acc[nt] = __builtin_amdgcn_wmma_f32_16x16x32_f16(
                          false, a, false, bm, (short)0, acc[nt], false, false);
        }
    }
    const int n  = lane & 15;
    const int mb = (lane & 16) ? 8 : 0;
    #pragma unroll
    for (int nt = 0; nt < 4; ++nt) {
        const int oc = nt * 16 + n;
        const float bs = bias[oc];
        #pragma unroll
        for (int r = 0; r < 8; ++r) {
            const int xo = xt * 16 + mb + r;
            out[((size_t)b * COUTc + oc) * HW + (size_t)y * Ww + xo] = acc[nt][r] + bs;
        }
    }
}

// ---------------------------------------------------------------------------
extern "C" void kernel_launch(void* const* d_in, const int* in_sizes, int n_in,
                              void* d_out, int out_size, void* d_ws, size_t ws_size,
                              hipStream_t stream) {
    const float* x      = (const float*)d_in[0];   // [8,64,128,128]
    const float* fea    = (const float*)d_in[1];   // [8,64,128,128]
    const float* weight = (const float*)d_in[2];   // [64,64,3,3]
    const float* bias   = (const float*)d_in[3];   // [64]
    const float* w_om   = (const float*)d_in[4];   // [216,64,3,3]
    const float* b_om   = (const float*)d_in[5];   // [216]
    float* out = (float*)d_out;                    // [8,64,128,128]

    char* ws = (char*)d_ws;
    const size_t SZ_IMG = (size_t)NB * HW * 64 * sizeof(float);     // 33.55 MB
    const size_t SZ_OM  = (size_t)NB * HW * COM * sizeof(float);    // 113.2 MB
    float*     xT   = (float*)ws;
    float*     feaT = (float*)(ws + SZ_IMG);
    float*     om   = (float*)(ws + 2 * SZ_IMG);
    _Float16*  wB1  = (_Float16*)(ws + 2 * SZ_IMG + SZ_OM);
    _Float16*  wB2  = wB1 + (size_t)NT1 * KSTEPS * 32 * 16;

    nchw_to_nhwc64<<<dim3(NB * 256, 2), 256, 0, stream>>>(x, fea, xT, feaT);

    const int packN = (NT1 + NT2) * KSTEPS * 32;                    // 10368
    pack_wmma_b<<<(packN + 255) / 256, 256, 0, stream>>>(w_om, weight, wB1, wB2);

    offset_conv_wmma<<<dim3(NB * 1024 / 4, NT1 / 2), 128, 0, stream>>>(feaT, b_om, wB1, om);

    deform_conv_wmma<<<NB * 1024 / 4, 128, 0, stream>>>(xT, om, bias, wB2, out);
}
